// Transformer_54357106098334
// MI455X (gfx1250) — matrix operand order repaired and checked by reading, not compiled, not run on hardware
//
#include <hip/hip_runtime.h>

typedef __attribute__((ext_vector_type(16))) __bf16 v16bf;
typedef __attribute__((ext_vector_type(8)))  float  v8f;

static constexpr int cL = 8, cH = 16, cE = 1024, cF = 4096, cV = 32000, cT = 2048;
static constexpr float cEPS = 1.1920928955078125e-07f;

union Frag16 { v16bf v; unsigned short u[16]; };

__device__ __forceinline__ unsigned short f32_bf16(float f) {
  unsigned int u = __float_as_uint(f);
  unsigned int r = u + 0x7FFFu + ((u >> 16) & 1u);   // round-to-nearest-even
  return (unsigned short)(r >> 16);
}

// ---------------------------------------------------------------- embedding
__global__ void embed_kernel(const int* __restrict__ tokens,
                             const float* __restrict__ table,
                             float* __restrict__ x) {
  const int t = blockIdx.x;
  const float* src = table + (long)tokens[t] * cE;
  float* dst = x + (long)t * cE;
  for (int i = threadIdx.x; i < cE; i += blockDim.x) dst[i] = src[i];
}

// ------------------------------------------------- RMSNorm (+opt residual)
__global__ void rmsnorm_kernel(const float* __restrict__ xin,
                               const float* __restrict__ resid,
                               const float* __restrict__ w,
                               float* __restrict__ xout,
                               unsigned short* __restrict__ xbf) {
  __shared__ float red[8];
  const int t = blockIdx.x;
  const long base = (long)t * cE;
  float vals[4];
  float ss = 0.f;
  #pragma unroll
  for (int j = 0; j < 4; ++j) {
    int i = threadIdx.x + j * 256;
    float v = xin[base + i];
    if (resid) v += resid[base + i];
    vals[j] = v;
    ss += v * v;
  }
  for (int off = 16; off >= 1; off >>= 1) ss += __shfl_xor(ss, off, 32);
  if ((threadIdx.x & 31) == 0) red[threadIdx.x >> 5] = ss;
  __syncthreads();
  float tot = 0.f;
  #pragma unroll
  for (int i = 0; i < 8; ++i) tot += red[i];
  const float rms = rsqrtf(tot / (float)cE + cEPS);
  #pragma unroll
  for (int j = 0; j < 4; ++j) {
    int i = threadIdx.x + j * 256;
    float o = vals[j] * rms * w[i];
    xout[base + i] = o;
    xbf[base + i] = f32_bf16(o);
  }
}

// --------------------------------------------------------- f32 -> bf16 copy
__global__ void cvt_bf16_kernel(const float* __restrict__ in,
                                unsigned short* __restrict__ out, long n) {
  long i = (long)blockIdx.x * blockDim.x + threadIdx.x;
  if (i < n) out[i] = f32_bf16(in[i]);
}

// --------------------------------------------------- RoPE (+ bf16 convert)
__global__ void rope_kernel(const float* __restrict__ in,
                            unsigned short* __restrict__ out) {
  const long idx = (long)blockIdx.x * blockDim.x + threadIdx.x; // H*T*64
  const int d = (int)(idx & 63);
  const int t = (int)((idx >> 6) & (cT - 1));
  const float x = in[idx];
  const float freq = __powf(10000.f, (-2.f / 64.f) * (float)(d & 31));
  const float ang = (float)t * freq;
  float s, c;
  __sincosf(ang, &s, &c);
  const float sgn = (d < 32) ? -1.f : 1.f;
  const float sw = in[idx + ((d < 32) ? 32 : -32)];
  out[idx] = f32_bf16(c * x + sgn * s * sw);
}

// --------------------------------------------------------------- bf16 GEMM
// C[M x N] = A_bf16[M x Kd] * B_f32[Kd x N] (+bias, opt ReLU/residual).
// 128x128 workgroup tile, 8 waves; wave = 2 row-blocks x 4 col-blocks =
// 8 x v_wmma_f32_16x16x32_bf16 per 32-deep k-step. A tile staged with
// GLOBAL_LOAD_ASYNC_TO_LDS_B128 (ASYNCcnt), B converted f32->bf16 into a
// column-major LDS tile so fragment reads are contiguous.
// HEADB: output column j belongs to head j/64 (B stride headB, C stride
// headC, per-head ldb/ldc) -- lets the per-head QKV GEMMs use the big tile.
// OUT_MODE: 0 = f32 store, 1 = bf16 store, 2 = f32 store + residual add.
template <int OUT_MODE, bool RELU, bool HEADB>
__global__ __launch_bounds__(256)
void gemm_bf16_kernel(const unsigned short* __restrict__ A, int lda,
                      const float* __restrict__ Bm, int ldb, long headB,
                      const float* __restrict__ bias,
                      void* __restrict__ Cv, int ldc, long headC,
                      const float* __restrict__ resid, int Kd) {
  __shared__ unsigned short As[128][32];   // [m][k] row-major bf16 (8KB)
  __shared__ unsigned short Bs[128][32];   // [n][k] col-major bf16 (8KB)
  const int m0 = blockIdx.y * 128;
  const int n0 = blockIdx.x * 128;
  const int tid = threadIdx.x;
  const int lane = tid & 31;
  const int wave = tid >> 5;
  const int rw = (wave & 3) * 32;          // wave row offset (2 row blocks)
  const int cw = (wave >> 2) * 64;         // wave col offset (4 col blocks)
  const int row = lane & 15;
  const int akb = (lane >= 16) ? 16 : 0;   // A-frag K base (contiguous K, matches B)
  const int bkb = (lane >= 16) ? 16 : 0;   // B-frag K base
  const int rowoff = (lane >= 16) ? 8 : 0; // C-frag row offset

  v8f acc[2][4] = {};

  for (int kt = 0; kt < Kd; kt += 32) {
    // ---- stage A asynchronously: 128x32 bf16 = 512 x 16B chunks
    {
      const unsigned short* Ag = A + (long)m0 * lda + kt;
      #pragma unroll
      for (int j = 0; j < 2; ++j) {
        const int i = tid + j * 256;           // chunk id 0..511
        const int r = i >> 2;                  // row 0..127
        const int c = (i & 3) * 8;             // 8 bf16 per 16B chunk
        const unsigned long long ga = (unsigned long long)(Ag + (long)r * lda + c);
        const unsigned lofs = (unsigned)(unsigned long long)(uintptr_t)&As[r][c];
        asm volatile("global_load_async_to_lds_b128 %0, %1, off"
                     :: "v"(lofs), "v"(ga) : "memory");
      }
    }
    // ---- stage B: f32 -> bf16, transpose to [n][k]
    for (int i = tid; i < 32 * 128; i += 256) {
      const int k = i >> 7, n = i & 127;
      const int colg = n0 + n;
      float f;
      if (HEADB)
        f = Bm[(long)(colg >> 6) * headB + (long)(kt + k) * ldb + (colg & 63)];
      else
        f = Bm[(long)(kt + k) * ldb + colg];
      Bs[n][k] = f32_bf16(f);
    }
    if (lane == 0 && kt + 32 < Kd && !HEADB)
      __builtin_prefetch(Bm + (long)(kt + 32) * ldb + n0, 0, 0);
    asm volatile("s_wait_asynccnt 0" ::: "memory");
    __syncthreads();

    // ---- fragments + 8 WMMAs per wave
    Frag16 af[2];
    #pragma unroll
    for (int rr = 0; rr < 2; ++rr) {
      const unsigned short* ar = &As[rw + rr * 16 + row][0];
      #pragma unroll
      for (int e = 0; e < 8; ++e) {
        af[rr].u[e] = ar[akb + e];
        af[rr].u[e + 8] = ar[akb + 8 + e];
      }
    }
    #pragma unroll
    for (int cc = 0; cc < 4; ++cc) {
      Frag16 bfrag;
      const unsigned short* br = &Bs[cw + cc * 16 + row][bkb];
      #pragma unroll
      for (int e = 0; e < 16; ++e) bfrag.u[e] = br[e];
      acc[0][cc] = __builtin_amdgcn_wmma_f32_16x16x32_bf16(
          false, af[0].v, false, bfrag.v, (short)0, acc[0][cc], false, false);
      acc[1][cc] = __builtin_amdgcn_wmma_f32_16x16x32_bf16(
          false, af[1].v, false, bfrag.v, (short)0, acc[1][cc], false, false);
    }
    __syncthreads();
  }

  // ---- epilogue
  float* Cf = (float*)Cv;
  unsigned short* Cb = (unsigned short*)Cv;
  #pragma unroll
  for (int cc = 0; cc < 4; ++cc) {
    const int colg = n0 + cw + cc * 16 + row;
    const float bi = bias ? bias[colg] : 0.f;
    #pragma unroll
    for (int rr = 0; rr < 2; ++rr) {
      #pragma unroll
      for (int r = 0; r < 8; ++r) {
        const int m = m0 + rw + rr * 16 + rowoff + r;
        float v = acc[rr][cc][r] + bi;
        if (RELU) v = fmaxf(v, 0.f);
        const long cidx = HEADB
            ? (long)(colg >> 6) * headC + (long)m * ldc + (colg & 63)
            : (long)m * ldc + colg;
        if (OUT_MODE == 2) v += resid[cidx];
        if (OUT_MODE == 1) Cb[cidx] = f32_bf16(v);
        else               Cf[cidx] = v;
      }
    }
  }
}

// -------------------------------------------------- flash attention (wave32)
// One wave per (head, 16-query block); streams 32-key tiles with online
// softmax. S = Q*K^T and O += P*V both via v_wmma_f32_16x16x32_bf16.
__global__ __launch_bounds__(32)
void attn_kernel(const unsigned short* __restrict__ Qh,
                 const unsigned short* __restrict__ Kh,
                 const unsigned short* __restrict__ Vh,
                 float* __restrict__ out /* [T][E], concat-head layout */) {
  __shared__ unsigned short Plds[16 * 32];   // P tile, C-layout -> A-layout hop
  __shared__ unsigned short Vlds[64 * 32];   // V tile [d][key] col-major
  const int h = blockIdx.y;
  const int qb = blockIdx.x;
  const int lane = threadIdx.x;
  const int row = lane & 15;
  const int akb = (lane >= 16) ? 16 : 0;
  const int bkb = (lane >= 16) ? 16 : 0;
  const int rowoff = (lane >= 16) ? 8 : 0;

  const unsigned short* Q = Qh + ((long)h * cT + qb * 16) * 64;
  const unsigned short* Kp = Kh + (long)h * cT * 64;
  const unsigned short* Vp = Vh + (long)h * cT * 64;

  Frag16 qf[2];
  #pragma unroll
  for (int c = 0; c < 2; ++c) {
    const unsigned short* qr = Q + row * 64 + c * 32;
    #pragma unroll
    for (int e = 0; e < 8; ++e) { qf[c].u[e] = qr[akb + e]; qf[c].u[e + 8] = qr[akb + 8 + e]; }
  }

  v8f o0 = {}, o1 = {}, o2 = {}, o3 = {};
  float mrow[8], lrow[8];
  #pragma unroll
  for (int r = 0; r < 8; ++r) { mrow[r] = -3.0e38f; lrow[r] = 0.f; }

  const int nkt = (qb * 16 + 15) / 32 + 1;
  for (int kt = 0; kt < nkt; ++kt) {
    const int k0 = kt * 32;
    // ---- S = Q (16x64) * K^T (64x32): 4 WMMAs -> two 16x16 C frags
    v8f s0 = {}, s1 = {};
    #pragma unroll
    for (int c = 0; c < 2; ++c) {
      Frag16 kf0, kf1;
      const unsigned short* kr0 = Kp + (long)(k0 + row) * 64 + c * 32 + bkb;
      const unsigned short* kr1 = kr0 + 16 * 64;
      #pragma unroll
      for (int e = 0; e < 16; ++e) { kf0.u[e] = kr0[e]; kf1.u[e] = kr1[e]; }
      s0 = __builtin_amdgcn_wmma_f32_16x16x32_bf16(false, qf[c].v, false, kf0.v,
                                                   (short)0, s0, false, false);
      s1 = __builtin_amdgcn_wmma_f32_16x16x32_bf16(false, qf[c].v, false, kf1.v,
                                                   (short)0, s1, false, false);
    }
    // ---- scale + causal mask + online softmax (row stats per 16-lane half)
    const int gc0 = k0 + row, gc1 = gc0 + 16;
    float p0[8], p1[8];
    #pragma unroll
    for (int r = 0; r < 8; ++r) {
      const int gr = qb * 16 + rowoff + r;
      p0[r] = (gc0 <= gr) ? s0[r] * 0.125f : -3.0e38f;
      p1[r] = (gc1 <= gr) ? s1[r] * 0.125f : -3.0e38f;
    }
    #pragma unroll
    for (int r = 0; r < 8; ++r) {
      float ml = fmaxf(p0[r], p1[r]);
      ml = fmaxf(ml, __shfl_xor(ml, 8, 32));
      ml = fmaxf(ml, __shfl_xor(ml, 4, 32));
      ml = fmaxf(ml, __shfl_xor(ml, 2, 32));
      ml = fmaxf(ml, __shfl_xor(ml, 1, 32));
      const float mnew = fmaxf(mrow[r], ml);
      const float corr = __expf(mrow[r] - mnew);
      mrow[r] = mnew;
      p0[r] = __expf(p0[r] - mnew);
      p1[r] = __expf(p1[r] - mnew);
      float rs = p0[r] + p1[r];
      rs += __shfl_xor(rs, 8, 32);
      rs += __shfl_xor(rs, 4, 32);
      rs += __shfl_xor(rs, 2, 32);
      rs += __shfl_xor(rs, 1, 32);
      lrow[r] = lrow[r] * corr + rs;
      o0[r] *= corr; o1[r] *= corr; o2[r] *= corr; o3[r] *= corr;
      Plds[(rowoff + r) * 32 + row]      = f32_bf16(p0[r]);
      Plds[(rowoff + r) * 32 + 16 + row] = f32_bf16(p1[r]);
    }
    // ---- stage V tile transposed: Vlds[d][key]
    for (int i = lane; i < 32 * 64; i += 32) {
      const int k = i >> 6, d = i & 63;
      Vlds[d * 32 + k] = Vp[(long)(k0 + k) * 64 + d];
    }
    __syncthreads();
    // ---- O += P (16x32) * V (32x64): 4 WMMAs
    Frag16 pf;
    const unsigned short* pr = &Plds[row * 32];
    #pragma unroll
    for (int e = 0; e < 8; ++e) { pf.u[e] = pr[akb + e]; pf.u[e + 8] = pr[akb + 8 + e]; }
    Frag16 vf0, vf1, vf2, vf3;
    #pragma unroll
    for (int e = 0; e < 16; ++e) {
      vf0.u[e] = Vlds[(0 * 16 + row) * 32 + bkb + e];
      vf1.u[e] = Vlds[(1 * 16 + row) * 32 + bkb + e];
      vf2.u[e] = Vlds[(2 * 16 + row) * 32 + bkb + e];
      vf3.u[e] = Vlds[(3 * 16 + row) * 32 + bkb + e];
    }
    o0 = __builtin_amdgcn_wmma_f32_16x16x32_bf16(false, pf.v, false, vf0.v, (short)0, o0, false, false);
    o1 = __builtin_amdgcn_wmma_f32_16x16x32_bf16(false, pf.v, false, vf1.v, (short)0, o1, false, false);
    o2 = __builtin_amdgcn_wmma_f32_16x16x32_bf16(false, pf.v, false, vf2.v, (short)0, o2, false, false);
    o3 = __builtin_amdgcn_wmma_f32_16x16x32_bf16(false, pf.v, false, vf3.v, (short)0, o3, false, false);
    __syncthreads();
  }
  // ---- normalize and store (concat heads: column h*64 + d)
  #pragma unroll
  for (int r = 0; r < 8; ++r) {
    const int gr = qb * 16 + rowoff + r;
    const float inv = 1.f / lrow[r];
    float* op = out + (long)gr * cE + h * 64;
    op[0 * 16 + row] = o0[r] * inv;
    op[1 * 16 + row] = o1[r] * inv;
    op[2 * 16 + row] = o2[r] * inv;
    op[3 * 16 + row] = o3[r] * inv;
  }
}

// ------------------------------------------------------------------- driver
extern "C" void kernel_launch(void* const* d_in, const int* in_sizes, int n_in,
                              void* d_out, int out_size, void* d_ws, size_t ws_size,
                              hipStream_t stream) {
  (void)in_sizes; (void)n_in; (void)out_size; (void)ws_size;
  const int*   tokens      = (const int*)d_in[0];
  const float* table       = (const float*)d_in[1];
  const float* attn_norm_w = (const float*)d_in[2];
  const float* Wq          = (const float*)d_in[3];
  const float* bq          = (const float*)d_in[4];
  const float* Wk          = (const float*)d_in[5];
  const float* bk          = (const float*)d_in[6];
  const float* Wv          = (const float*)d_in[7];
  const float* bv          = (const float*)d_in[8];
  const float* mlp_norm_w  = (const float*)d_in[9];
  const float* W1          = (const float*)d_in[10];
  const float* b1          = (const float*)d_in[11];
  const float* W2          = (const float*)d_in[12];
  const float* b2          = (const float*)d_in[13];
  const float* Wp          = (const float*)d_in[14];
  const float* bp          = (const float*)d_in[15];
  float* logits = (float*)d_out;

  // workspace carve (~64 MB, all 4-byte aligned)
  char* w = (char*)d_ws;
  float*          x    = (float*)w;          w += (size_t)cT * cE * 4;
  unsigned short* xn   = (unsigned short*)w; w += (size_t)cT * cE * 2;
  float*          qf   = (float*)w;          w += (size_t)cH * cT * 64 * 4;
  float*          kf   = (float*)w;          w += (size_t)cH * cT * 64 * 4;
  unsigned short* q16  = (unsigned short*)w; w += (size_t)cH * cT * 64 * 2;
  unsigned short* k16  = (unsigned short*)w; w += (size_t)cH * cT * 64 * 2;
  unsigned short* v16  = (unsigned short*)w; w += (size_t)cH * cT * 64 * 2;
  float*          attn = (float*)w;          w += (size_t)cT * cE * 4;
  unsigned short* hbuf = (unsigned short*)w; w += (size_t)cT * cF * 2;

  embed_kernel<<<cT, 256, 0, stream>>>(tokens, table, x);

  const long headB = (long)cE * 64;   // per-head B stride for QKV
  const long headC = (long)cT * 64;   // per-head C stride ([H][T][64])

  for (int l = 0; l < cL; ++l) {
    rmsnorm_kernel<<<cT, 256, 0, stream>>>(x, nullptr, attn_norm_w + (long)l * cE, x, xn);

    const long wqkv = (long)l * cH * cE * 64;
    gemm_bf16_kernel<0, false, true><<<dim3(cH * 64 / 128, cT / 128, 1), 256, 0, stream>>>(
        xn, cE, Wq + wqkv, 64, headB, bq + (long)l * cH * 64,
        qf, 64, headC, nullptr, cE);
    gemm_bf16_kernel<0, false, true><<<dim3(cH * 64 / 128, cT / 128, 1), 256, 0, stream>>>(
        xn, cE, Wk + wqkv, 64, headB, bk + (long)l * cH * 64,
        kf, 64, headC, nullptr, cE);
    gemm_bf16_kernel<1, false, true><<<dim3(cH * 64 / 128, cT / 128, 1), 256, 0, stream>>>(
        xn, cE, Wv + wqkv, 64, headB, bv + (long)l * cH * 64,
        v16, 64, headC, nullptr, cE);

    rope_kernel<<<(cH * cT * 64) / 256, 256, 0, stream>>>(qf, q16);
    rope_kernel<<<(cH * cT * 64) / 256, 256, 0, stream>>>(kf, k16);

    attn_kernel<<<dim3(cT / 16, cH), 32, 0, stream>>>(q16, k16, v16, attn);

    rmsnorm_kernel<<<cT, 256, 0, stream>>>(x, attn, mlp_norm_w + (long)l * cE, x, xn);

    gemm_bf16_kernel<1, true, false><<<dim3(cF / 128, cT / 128, 1), 256, 0, stream>>>(
        xn, cE, W1 + (long)l * cE * cF, cF, 0, b1 + (long)l * cF,
        hbuf, cF, 0, nullptr, cE);
    gemm_bf16_kernel<2, false, false><<<dim3(cE / 128, cT / 128, 1), 256, 0, stream>>>(
        hbuf, cF, W2 + (long)l * cF * cE, cE, 0, b2 + (long)l * cE,
        x, cE, 0, x, cF);
  }

  cvt_bf16_kernel<<<(cT * cE) / 256, 256, 0, stream>>>(x, xn, (long)cT * cE);
  gemm_bf16_kernel<0, false, false><<<dim3(cV / 128, cT / 128, 1), 256, 0, stream>>>(
      xn, cE, Wp, cV, 0, bp, logits, cV, 0, nullptr, cE);
}
